// BernoulliEdge_721554505987
// MI455X (gfx1250) — matrix operand order repaired and checked by reading, not compiled, not run on hardware
//
#include <hip/hip_runtime.h>
#include <hip/hip_bf16.h>

#define B_  32
#define N_  1024
#define F_  512
#define NN_ (N_ * N_)

typedef __attribute__((ext_vector_type(16))) __bf16 v16bf;
typedef __attribute__((ext_vector_type(8)))  __bf16 v8bf;
typedef __attribute__((ext_vector_type(8)))  float  v8f;
typedef __attribute__((ext_vector_type(4)))  float  v4f;   // clang vector: OK for NT builtins

__device__ __forceinline__ float sigmoidf_(float x) { return 1.0f / (1.0f + __expf(-x)); }

// ---------------- counter-based RNG (deterministic, graph-replay safe) -------
__device__ __forceinline__ unsigned hash_u32(unsigned x) {
  x ^= x >> 16; x *= 0x7feb352dU;
  x ^= x >> 15; x *= 0x846ca68bU;
  x ^= x >> 16; return x;
}
__device__ __forceinline__ float uniform_open(unsigned idx, unsigned stream) {
  unsigned h = hash_u32(idx ^ (stream * 0x9E3779B9U + 0x85ebca6bU));
  h = hash_u32(h + (stream << 3) + 0x1b873593U);
  float u = (float)(h >> 8) * (1.0f / 16777216.0f);   // [0,1), 24-bit mantissa
  return 1e-6f + u * (1.0f - 2e-6f);                  // (1e-6, 1-1e-6) like jax.uniform
}

// ---------------- Phase A: leftc[b,f] = nodes[b, nn_b, :] @ W1[0:F, f] + b1[f]
__global__ __launch_bounds__(256) void leftc_kernel(
    const float* __restrict__ nodes, const int* __restrict__ num_nodes,
    const float* __restrict__ W1, const float* __restrict__ b1,
    float* __restrict__ leftc) {
  int gid = blockIdx.x * 256 + threadIdx.x;            // B_*F_ threads
  int b = gid >> 9;
  int f = gid & (F_ - 1);
  int nn = num_nodes[b];
  const float* lrow = nodes + ((size_t)b * N_ + nn) * F_;
  float a0 = 0.f, a1 = 0.f, a2 = 0.f, a3 = 0.f;        // break the serial FMA chain
  for (int k = 0; k < F_; k += 4) {
    a0 = fmaf(lrow[k + 0], W1[(size_t)(k + 0) * F_ + f], a0);
    a1 = fmaf(lrow[k + 1], W1[(size_t)(k + 1) * F_ + f], a1);
    a2 = fmaf(lrow[k + 2], W1[(size_t)(k + 2) * F_ + f], a2);
    a3 = fmaf(lrow[k + 3], W1[(size_t)(k + 3) * F_ + f], a3);
  }
  leftc[gid] = b1[f] + ((a0 + a1) + (a2 + a3));
}

// ---------------- Phase A2: W1bT[f][k] = bf16(W1[F_+k, f])  (bf16, transposed)
__global__ __launch_bounds__(256) void w1bt_kernel(
    const float* __restrict__ W1, unsigned short* __restrict__ w1bt) {
  int gid = blockIdx.x * 256 + threadIdx.x;            // F_*F_ threads
  int f = gid >> 9;
  int k = gid & (F_ - 1);
  __bf16 h = (__bf16)W1[(size_t)(F_ + k) * F_ + f];
  w1bt[(size_t)f * F_ + k] = __builtin_bit_cast(unsigned short, h);
}

// ---------------- Phase A3: pre-swizzled bf16 A fragments of `nodes`
// Group g = ((rowblock*16 + ks)*32 + lane); 16 bf16 per group = one lane's
// WMMA A-fragment for k-step ks (contiguous k = kh*16 + s order baked in).
__global__ __launch_bounds__(256) void nodesA_kernel(
    const float* __restrict__ nodes, unsigned short* __restrict__ out) {
  const int g    = blockIdx.x * 256 + threadIdx.x;     // 2^20 groups
  const int rb   = g >> 9;                             // 16-row block (0..2047)
  const int ks   = (g >> 5) & 15;                      // k-step (32 K each)
  const int lane = g & 31;
  const int m    = lane & 15;
  const int kh   = lane >> 4;
  const size_t base = (size_t)(rb * 16 + m) * F_ + ks * 32 + kh * 16;
  unsigned short* dst = out + (size_t)g * 16;
#pragma unroll
  for (int j = 0; j < 8; ++j) {
    dst[j]     = __builtin_bit_cast(unsigned short, (__bf16)nodes[base + j]);
    dst[8 + j] = __builtin_bit_cast(unsigned short, (__bf16)nodes[base + 8 + j]);
  }
}

// ---------------- Phase B: fused  probs = clip(sigmoid(leaky(A@W1b + leftc) @ W2 + b2))
// 128 threads = 4 waves; block = 64 rows. A fragments live in VGPRs (coalesced
// loads from pre-swizzled nodesA); B chunk staged in 64KB LDS, XOR-swizzled.
__global__ __launch_bounds__(128) void mlp_probs_kernel(
    const unsigned short* __restrict__ nodesA_u, const float* __restrict__ leftc,
    const unsigned short* __restrict__ w1bt_u, const float* __restrict__ W2,
    const float* __restrict__ b2, float* __restrict__ probs) {
  __shared__ __bf16 sB[64 * F_];                       // 64 KB: 64 cols x 512 K
  const __bf16* __restrict__ nodesA = (const __bf16*)nodesA_u;
  const __bf16* __restrict__ w1bt   = (const __bf16*)w1bt_u;

  const int tid  = threadIdx.x;
  const int wave = tid >> 5;
  const int lane = tid & 31;
  const int m    = lane & 15;                          // row (A) / col (B,C) within tile
  const int kh   = lane >> 4;                          // K-half selector
  const int R0   = blockIdx.x * 64;                    // global row base (row = b*N + n)
  const int b    = R0 >> 10;
  const int rb   = (R0 >> 4) + wave;                   // this wave's 16-row block

  // All 16 A fragments: one fully-coalesced 32B read per lane per fragment.
  v16bf afr[16];
#pragma unroll
  for (int ks = 0; ks < 16; ++ks) {
    const v8bf* pa = (const v8bf*)(nodesA + ((size_t)(rb * 16 + ks) * 32 + lane) * 16);
    afr[ks] = __builtin_shufflevector(pa[0], pa[1],
        0, 1, 2, 3, 4, 5, 6, 7, 8, 9, 10, 11, 12, 13, 14, 15);
  }

  float rowpart[8];
  for (int i = 0; i < 8; ++i) rowpart[i] = 0.0f;

  for (int nc = 0; nc < 8; ++nc) {                     // 8 chunks of 64 output columns
    const int colBase = nc * 64;

    // Cooperative stage of B chunk (64 cols x 512 K bf16) into LDS.
    // 32B chunks; dest chunk index XOR-swizzled by column to spread banks.
    __syncthreads();
    for (int idx = tid; idx < 64 * 32; idx += 128) {
      const int col = idx >> 5;                        // local column 0..63
      const int c   = idx & 31;                        // 32B chunk within column
      const v8bf* src = (const v8bf*)(w1bt + (size_t)(colBase + col) * F_ + c * 16);
      const v8bf s0 = src[0], s1 = src[1];
      v8bf* dst = (v8bf*)&sB[col * F_ + ((c ^ (col & 31)) << 4)];
      dst[0] = s0;
      dst[1] = s1;
    }
    __syncthreads();

    float w2v[4];
    v8f acc[4];
    for (int t = 0; t < 4; ++t) {
      const int col = colBase + t * 16 + m;
      const float lc = leftc[b * F_ + col];            // includes b1; row-independent
      w2v[t] = W2[col];
      for (int i = 0; i < 8; ++i) acc[t][i] = lc;
    }

#pragma unroll
    for (int s = 0; s < 64; ++s) {
      const int t    = s & 3;
      const int ks   = s >> 2;
      const int colL = t * 16 + m;                     // local column of this lane
      const int c    = ks * 2 + kh;                    // 32B chunk = 16 K values
      const v8bf* pb = (const v8bf*)&sB[colL * F_ + ((c ^ (colL & 31)) << 4)];
      const v16bf bfrag = __builtin_shufflevector(pb[0], pb[1],
          0, 1, 2, 3, 4, 5, 6, 7, 8, 9, 10, 11, 12, 13, 14, 15);
      acc[t] = __builtin_amdgcn_wmma_f32_16x16x32_bf16(
          false, afr[ks], false, bfrag, (short)0, acc[t], false, false);
    }

    // fuse: h = leaky_relu(acc); rowpart += h * W2[col]
    for (int t = 0; t < 4; ++t)
      for (int i = 0; i < 8; ++i) {
        float v = acc[t][i];
        v = (v > 0.0f) ? v : 0.01f * v;
        rowpart[i] = fmaf(v, w2v[t], rowpart[i]);
      }
  }
  // C layout: VGPR i, lanes 0-15 -> row i, lanes 16-31 -> row i+8; reduce 16 cols.
  for (int msk = 8; msk >= 1; msk >>= 1)
    for (int i = 0; i < 8; ++i)
      rowpart[i] += __shfl_xor(rowpart[i], msk, 32);
  if (m == 0) {
    const float bb2 = b2[0];
    for (int i = 0; i < 8; ++i) {
      const int row = R0 + (wave << 4) + i + kh * 8;   // == b*N + n
      float p = sigmoidf_(rowpart[i] + bb2);
      p = fminf(fmaxf(p, 1e-4f), 1.0f - 1e-4f);
      probs[row] = p;
    }
  }
}

// ---------------- Phase C: row/col substitution + logistic sample + gumbel hard
__global__ __launch_bounds__(256) void edge_sample_kernel(
    const float* __restrict__ weights, const int* __restrict__ num_nodes,
    const float* __restrict__ probs, float* __restrict__ adj_out,
    float* __restrict__ w_out) {
  const unsigned gid  = blockIdx.x * 256 + threadIdx.x;
  const unsigned idx4 = gid * 4u;                      // 4 consecutive j per thread
  const int b   = idx4 >> 20;                          // N*N = 2^20
  const unsigned rem = idx4 & (NN_ - 1);
  const int i   = rem >> 10;
  const int j0  = rem & (N_ - 1);
  const int nn  = num_nodes[b];

  // streaming read: non-temporal so the 134MB weights pass doesn't evict L2
  const v4f w4 = __builtin_nontemporal_load((const v4f*)(weights + (size_t)idx4));
  float wv[4] = {w4.x, w4.y, w4.z, w4.w};
  float av[4];

  const float pi_row = probs[b * N_ + i];              // used when j == nn && i < nn
  for (int e = 0; e < 4; ++e) {
    const int j = j0 + e;
    float w = wv[e];
    if (i == nn && j < nn) w = probs[b * N_ + j];      // row update
    if (j == nn && i < nn) w = pi_row;                 // column update
    wv[e] = w;

    const unsigned idx = idx4 + e;
    // sample = sigmoid(logit(e) + logit(w)), e ~ U(1e-6, 1-1e-6)
    const float eu = uniform_open(idx, 1u);
    const float s  = sigmoidf_(logf(eu) - log1pf(-eu) + logf(w) - log1pf(-w));
    // gumbel-softmax hard over {1-s, s}
    const float s1 = fminf(fmaxf(s, 1e-4f), 1.0f - 1e-4f);
    const float s0 = fminf(fmaxf(1.0f - s, 1e-4f), 1.0f - 1e-4f);
    const float u0 = uniform_open(idx, 2u);
    const float u1 = uniform_open(idx, 3u);
    const float a0 = logf(s0) - logf(-logf(u0));
    const float a1 = logf(s1) - logf(-logf(u1));
    const float y1 = sigmoidf_(a1 - a0);
    const float hard = (a1 > a0) ? 1.0f : 0.0f;
    av[e] = (hard + y1) - y1;                          // matches hard + y1 - stop_grad(y1)
  }
  // streaming writes: non-temporal (one-touch 268MB total)
  v4f ao; ao.x = av[0]; ao.y = av[1]; ao.z = av[2]; ao.w = av[3];
  v4f wo; wo.x = wv[0]; wo.y = wv[1]; wo.z = wv[2]; wo.w = wv[3];
  __builtin_nontemporal_store(ao, (v4f*)(adj_out + (size_t)idx4));
  __builtin_nontemporal_store(wo, (v4f*)(w_out + (size_t)idx4));
}

// ---------------- launcher ---------------------------------------------------
extern "C" void kernel_launch(void* const* d_in, const int* in_sizes, int n_in,
                              void* d_out, int out_size, void* d_ws, size_t ws_size,
                              hipStream_t stream) {
  const float* nodes     = (const float*)d_in[0];
  /* d_in[1] = adj (zeros, unused) */
  const float* weights   = (const float*)d_in[2];
  const int*   num_nodes = (const int*)d_in[3];
  /* d_in[4] = B scalar (compile-time constant here) */
  const float* W1        = (const float*)d_in[5];
  const float* b1        = (const float*)d_in[6];
  const float* W2        = (const float*)d_in[7];
  const float* b2        = (const float*)d_in[8];

  float* adj_out = (float*)d_out;
  float* w_out   = (float*)d_out + (size_t)B_ * NN_;

  char* ws = (char*)d_ws;
  float*          leftc  = (float*)ws;                          // 32*512*4  =  64 KB
  unsigned short* w1bt   = (unsigned short*)(ws + 64 * 1024);   // 512*512*2 = 512 KB
  float*          probs  = (float*)(ws + 576 * 1024);           // 32*1024*4 = 128 KB
  unsigned short* nodesA = (unsigned short*)(ws + 704 * 1024);  // 16M*2     =  32 MB

  leftc_kernel<<<(B_ * F_) / 256, 256, 0, stream>>>(nodes, num_nodes, W1, b1, leftc);
  w1bt_kernel<<<(F_ * F_) / 256, 256, 0, stream>>>(W1, w1bt);
  nodesA_kernel<<<(B_ * N_ * F_ / 16) / 256, 256, 0, stream>>>(nodes, nodesA);
  mlp_probs_kernel<<<(B_ * N_) / 64, 128, 0, stream>>>(nodesA, leftc, w1bt, W2, b2, probs);
  edge_sample_kernel<<<(B_ * NN_) / (4 * 256), 256, 0, stream>>>(
      weights, num_nodes, probs, adj_out, w_out);
}
